// MultiHeadAttention_28982439313762
// MI455X (gfx1250) — matrix operand order repaired and checked by reading, not compiled, not run on hardware
//
#include <hip/hip_runtime.h>
#include <hip/hip_bf16.h>

#define D_MODEL 1024
#define HEADS   16
#define DK      64
#define BS      4
#define SEQ     2048

typedef __attribute__((ext_vector_type(16))) __bf16 v16bf;
typedef __attribute__((ext_vector_type(8)))  float  v8f;

union Frag { v16bf v; unsigned int u[8]; __bf16 h[16]; };
union Acc  { v8f  v; float f[8]; };

__device__ inline unsigned int pack_bf2(float a, float b) {
  union { __bf16 h[2]; unsigned int u; } x;
  x.h[0] = (__bf16)a; x.h[1] = (__bf16)b;
  return x.u;
}

// LDS byte offset of a __shared__ object: low 32 bits of the generic address
// (flat LDS aperture truncation rule, ISA 10.2).
__device__ inline unsigned lds_off(const void* p) { return (unsigned)(size_t)p; }

// Async global->LDS copy, 4 bytes per lane (GLOBAL_LOAD_ASYNC_TO_LDS_B32,
// tracked by ASYNCcnt).  VDST = per-lane LDS byte address, VADDR = 64-bit
// global address (GV mode).
#define ASYNC_G2L_B32(ldsoff, gaddr)                                          \
  asm volatile("global_load_async_to_lds_b32 %0, %1, off"                     \
               :: "v"(ldsoff), "v"(gaddr) : "memory")
#define WAIT_ASYNC0() asm volatile("s_wait_asynccnt 0" ::: "memory")

// ---------------------------------------------------------------------------
// GEMM: Y = X @ W^T + bias.  M=8192, N=1024, K=1024.
// Workgroup tile: 128(M) x 64(N), 8 waves each own a 16x64 strip.
// IN_BF16: 0 = fp32 input (convert in VGPRs), 1 = bf16 input (async to LDS).
// OUT_MODE: 0 = bf16 head-split [b,h,s,d]
//           1 = bf16 head-split transposed [b,h,d,s]   (for K^T)
//           2 = fp32 row-major [m,e]                   (final output)
// ---------------------------------------------------------------------------
template<int IN_BF16, int OUT_MODE>
__global__ __launch_bounds__(256) void gemm_bias_kernel(
    const void* __restrict__ Xv, const float* __restrict__ W,
    const float* __restrict__ bias, void* __restrict__ Yv)
{
  __shared__ __align__(16) __bf16 Aa[128][34];   // [m][k] bf16, padded
  __shared__ __align__(16) __bf16 Bt[32][66];    // [k][n] bf16 (W transposed), padded

  const int t    = threadIdx.x;
  const int lane = t & 31;
  const int w    = t >> 5;
  const int lrow = lane & 15;     // fragment row / column-in-subtile
  const int kg   = lane >> 4;     // K-group (A) / M-group (C,D)
  const int m0   = blockIdx.x * 128;
  const int n0   = blockIdx.y * 64;

  Acc acc[4];
#pragma unroll
  for (int i = 0; i < 4; i++)
#pragma unroll
    for (int r = 0; r < 8; r++) acc[i].f[r] = 0.f;

  for (int k0 = 0; k0 < D_MODEL; k0 += 32) {
    // ---- stage A tile (128 x 32) into LDS as bf16 ----
    if (IN_BF16) {
      const unsigned int* Xu = (const unsigned int*)Xv;   // 2 bf16 per uint
#pragma unroll
      for (int i = 0; i < 8; i++) {
        int flat = t + i * 256;                 // 0..2047 bf16-pairs
        int row = flat >> 4, cp = flat & 15;
        const unsigned int* gp =
            &Xu[(size_t)(m0 + row) * (D_MODEL / 2) + (k0 >> 1) + cp];
        ASYNC_G2L_B32(lds_off(&Aa[row][2 * cp]),
                      (unsigned long long)(size_t)gp);
      }
    } else {
      const float* X = (const float*)Xv;
#pragma unroll
      for (int i = 0; i < 8; i++) {
        int flat = t + i * 256;
        int row = flat >> 4, cp = flat & 15;
        const float* p = &X[(size_t)(m0 + row) * D_MODEL + k0 + 2 * cp];
        float2 f = *(const float2*)p;
        *(unsigned int*)&Aa[row][2 * cp] = pack_bf2(f.x, f.y);
      }
    }
    // ---- stage W tile (64 x 32), transposed into Bt[k][n] ----
#pragma unroll
    for (int i = 0; i < 2; i++) {
      int flat = t + i * 256;                   // 0..511 float4's
      int e = flat >> 3, c4 = (flat & 7) * 4;
      const float4 f = *(const float4*)&W[(size_t)(n0 + e) * D_MODEL + k0 + c4];
      Bt[c4 + 0][e] = (__bf16)f.x;
      Bt[c4 + 1][e] = (__bf16)f.y;
      Bt[c4 + 2][e] = (__bf16)f.z;
      Bt[c4 + 3][e] = (__bf16)f.w;
    }
    if (k0 + 32 < D_MODEL) {
      // prefetch next A stripe (-> global_prefetch_b8)
      if (IN_BF16)
        __builtin_prefetch((const char*)Xv + ((size_t)(m0 + (t >> 1)) * D_MODEL + k0 + 32) * 2, 0, 1);
      else
        __builtin_prefetch((const char*)Xv + ((size_t)(m0 + (t >> 1)) * D_MODEL + k0 + 32) * 4, 0, 1);
    }
    if (IN_BF16) WAIT_ASYNC0();
    __syncthreads();

    // ---- A fragment: 16x32 bf16, lane=row, K-groups of 8 ----
    Frag a;
    {
      const int arow = w * 16 + lrow;
      const int kb = kg * 8;
#pragma unroll
      for (int r = 0; r < 4; r++) a.u[r]     = *(const unsigned int*)&Aa[arow][kb + 2 * r];
#pragma unroll
      for (int r = 0; r < 4; r++) a.u[4 + r] = *(const unsigned int*)&Aa[arow][16 + kb + 2 * r];
    }
    // ---- 4 N-subtiles ----
#pragma unroll
    for (int ns = 0; ns < 4; ns++) {
      Frag bf;                                   // B: lane = N column, halves = K groups
#pragma unroll
      for (int j = 0; j < 16; j++) bf.h[j] = Bt[(j < 8 ? kg * 8 + j : 16 + kg * 8 + j - 8)][ns * 16 + lrow];
      acc[ns].v = __builtin_amdgcn_wmma_f32_16x16x32_bf16(
          false, a.v, false, bf.v, (short)0, acc[ns].v, false, false);
    }
    __syncthreads();
  }

  // ---- bias + writeback ----
#pragma unroll
  for (int ns = 0; ns < 4; ns++) {
    const int e = n0 + ns * 16 + lrow;
    const float bv = bias[e];
#pragma unroll
    for (int r = 0; r < 8; r++) {
      const int m = m0 + w * 16 + kg * 8 + r;
      const float val = acc[ns].f[r] + bv;
      if (OUT_MODE == 2) {
        ((float*)Yv)[(size_t)m * D_MODEL + e] = val;
      } else {
        const int b = m >> 11, s = m & (SEQ - 1);
        const int hh = e >> 6, d = e & 63;
        __bf16* Y = (__bf16*)Yv;
        size_t idx;
        if (OUT_MODE == 0) idx = (((size_t)b * HEADS + hh) * SEQ + s) * DK + d;
        else               idx = (((size_t)b * HEADS + hh) * DK + d) * SEQ + s;
        Y[idx] = (__bf16)val;
      }
    }
  }
}

// ---------------------------------------------------------------------------
// Flash attention: per workgroup = (b, h, 128 q-rows).  8 waves x 16 rows.
// K loop over 32-wide key blocks; tiles staged with async global->LDS copies;
// online softmax; everything bf16 WMMA.
// ---------------------------------------------------------------------------
__global__ __launch_bounds__(256) void flash_attn_kernel(
    const __bf16* __restrict__ Qh, const __bf16* __restrict__ KhT,
    const __bf16* __restrict__ Vh, const int* __restrict__ mask,
    __bf16* __restrict__ Oc)
{
  __shared__ __align__(16) __bf16 Kt[64][34];        // [d][kcol] = K^T tile
  __shared__ __align__(16) __bf16 Vt[32][66];        // [k][d]    = V tile
  __shared__ __align__(16) __bf16 Pt[8][16][34];     // per-wave P scratch

  const int t    = threadIdx.x;
  const int lane = t & 31;
  const int w    = t >> 5;
  const int lrow = lane & 15;
  const int kg   = lane >> 4;
  const int b    = blockIdx.z, h = blockIdx.y;
  const int qbase = blockIdx.x * 128;
  const size_t bh = (size_t)b * HEADS + h;

  // Q fragments (d=0..31 and d=32..63), pre-scaled by 1/sqrt(DK)=0.125
  Frag aq0, aq1;
  {
    const unsigned int* qrow =
        (const unsigned int*)(Qh + (bh * SEQ + qbase + w * 16 + lrow) * DK);
#pragma unroll
    for (int r = 0; r < 4; r++) {
      aq0.u[r]     = qrow[kg * 4 + r];
      aq0.u[4 + r] = qrow[8 + kg * 4 + r];
      aq1.u[r]     = qrow[16 + kg * 4 + r];
      aq1.u[4 + r] = qrow[24 + kg * 4 + r];
    }
#pragma unroll
    for (int i = 0; i < 16; i++) {
      aq0.h[i] = (__bf16)((float)aq0.h[i] * 0.125f);
      aq1.h[i] = (__bf16)((float)aq1.h[i] * 0.125f);
    }
  }

  Acc o[4];
#pragma unroll
  for (int i = 0; i < 4; i++)
#pragma unroll
    for (int r = 0; r < 8; r++) o[i].f[r] = 0.f;
  float m_run[8], l_run[8];
#pragma unroll
  for (int r = 0; r < 8; r++) { m_run[r] = -3.0e38f; l_run[r] = 0.f; }

  const int* mrow = mask + (size_t)b * SEQ * SEQ;    // mask[b][0][:][:]

  for (int kt = 0; kt < SEQ / 32; kt++) {
    const int kc0 = kt * 32;
    // ---- stage K^T tile (64 d x 32 kcols) and V tile, async to LDS ----
    {
      const unsigned int* src = (const unsigned int*)KhT;
#pragma unroll
      for (int i = 0; i < 4; i++) {
        int flat = t + i * 256;                 // 0..1023 uints
        int d = flat >> 4, cu = flat & 15;
        const unsigned int* gp = &src[(bh * DK + d) * (SEQ / 2) + (kc0 >> 1) + cu];
        ASYNC_G2L_B32(lds_off(&Kt[d][2 * cu]), (unsigned long long)(size_t)gp);
      }
      const unsigned int* vs = (const unsigned int*)Vh;
#pragma unroll
      for (int i = 0; i < 4; i++) {
        int flat = t + i * 256;
        int kk = flat >> 5, cu = flat & 31;
        const unsigned int* gp = &vs[(bh * SEQ + kc0 + kk) * (DK / 2) + cu];
        ASYNC_G2L_B32(lds_off(&Vt[kk][2 * cu]), (unsigned long long)(size_t)gp);
      }
    }
    WAIT_ASYNC0();
    __syncthreads();

    // ---- scores: S(16x32) = Qw(16x64) @ K^T(64x32), two 16-col subtiles ----
    Acc sA[2];
#pragma unroll
    for (int ns = 0; ns < 2; ns++) {
      Frag bk0, bk1;
#pragma unroll
      for (int j = 0; j < 16; j++) {
        const int kk = (j < 8 ? kg * 8 + j : 16 + kg * 8 + j - 8);
        bk0.h[j] = Kt[kk][ns * 16 + lrow];       // d=0..31
        bk1.h[j] = Kt[32 + kk][ns * 16 + lrow];  // d=32..63
      }
      Acc z;
#pragma unroll
      for (int r = 0; r < 8; r++) z.f[r] = 0.f;
      z.v = __builtin_amdgcn_wmma_f32_16x16x32_bf16(false, aq0.v, false, bk0.v,
                                                    (short)0, z.v, false, false);
      z.v = __builtin_amdgcn_wmma_f32_16x16x32_bf16(false, aq1.v, false, bk1.v,
                                                    (short)0, z.v, false, false);
      sA[ns] = z;
    }
    // ---- mask ----
#pragma unroll
    for (int ns = 0; ns < 2; ns++) {
      const int kcol = kc0 + ns * 16 + lrow;
#pragma unroll
      for (int r = 0; r < 8; r++) {
        const int q = qbase + w * 16 + kg * 8 + r;
        const int mk = mrow[(size_t)q * SEQ + kcol];
        sA[ns].f[r] = mk ? sA[ns].f[r] : -1.0e9f;
      }
    }
    // ---- online softmax (row = reduction across the 16 lanes of the group) ----
    float corr[8];
#pragma unroll
    for (int r = 0; r < 8; r++) {
      float mx = fmaxf(sA[0].f[r], sA[1].f[r]);
#pragma unroll
      for (int d = 1; d < 16; d <<= 1) mx = fmaxf(mx, __shfl_xor(mx, d, 32));
      const float mnew = fmaxf(m_run[r], mx);
      corr[r] = __expf(m_run[r] - mnew);
      m_run[r] = mnew;
      sA[0].f[r] = __expf(sA[0].f[r] - mnew);
      sA[1].f[r] = __expf(sA[1].f[r] - mnew);
      float rs = sA[0].f[r] + sA[1].f[r];
#pragma unroll
      for (int d = 1; d < 16; d <<= 1) rs += __shfl_xor(rs, d, 32);
      l_run[r] = l_run[r] * corr[r] + rs;
    }
#pragma unroll
    for (int ns = 0; ns < 4; ns++)
#pragma unroll
      for (int r = 0; r < 8; r++) o[ns].f[r] *= corr[r];

    // ---- P: D-layout -> A-fragment layout via per-wave LDS ----
#pragma unroll
    for (int ns = 0; ns < 2; ns++)
#pragma unroll
      for (int r = 0; r < 8; r++)
        Pt[w][kg * 8 + r][ns * 16 + lrow] = (__bf16)sA[ns].f[r];
    asm volatile("s_wait_dscnt 0" ::: "memory");
    Frag pa;
    {
      const int kb = kg * 8;
#pragma unroll
      for (int r = 0; r < 4; r++) pa.u[r]     = *(const unsigned int*)&Pt[w][lrow][kb + 2 * r];
#pragma unroll
      for (int r = 0; r < 4; r++) pa.u[4 + r] = *(const unsigned int*)&Pt[w][lrow][16 + kb + 2 * r];
    }
    // ---- O(16x64) += P(16x32) @ V(32x64) ----
#pragma unroll
    for (int ns = 0; ns < 4; ns++) {
      Frag bv;
#pragma unroll
      for (int j = 0; j < 16; j++) bv.h[j] = Vt[(j < 8 ? kg * 8 + j : 16 + kg * 8 + j - 8)][ns * 16 + lrow];
      o[ns].v = __builtin_amdgcn_wmma_f32_16x16x32_bf16(
          false, pa.v, false, bv.v, (short)0, o[ns].v, false, false);
    }
    __syncthreads();
  }

  // ---- normalize and write concat layout Oc[b][s][h*64+d] ----
#pragma unroll
  for (int r = 0; r < 8; r++) {
    const float inv = 1.0f / l_run[r];
#pragma unroll
    for (int ns = 0; ns < 4; ns++) o[ns].f[r] *= inv;
  }
#pragma unroll
  for (int ns = 0; ns < 4; ns++) {
    const int d = ns * 16 + lrow;
#pragma unroll
    for (int r = 0; r < 8; r++) {
      const int s = qbase + w * 16 + kg * 8 + r;
      Oc[((size_t)b * SEQ + s) * D_MODEL + h * DK + d] = (__bf16)o[ns].f[r];
    }
  }
}

// ---------------------------------------------------------------------------
extern "C" void kernel_launch(void* const* d_in, const int* in_sizes, int n_in,
                              void* d_out, int out_size, void* d_ws, size_t ws_size,
                              hipStream_t stream) {
  const float* q    = (const float*)d_in[0];
  const float* k    = (const float*)d_in[1];
  const float* v    = (const float*)d_in[2];
  const int*   mask = (const int*)d_in[3];
  const float* Wq   = (const float*)d_in[4];
  const float* bq   = (const float*)d_in[5];
  const float* Wk   = (const float*)d_in[6];
  const float* bk   = (const float*)d_in[7];
  const float* Wv   = (const float*)d_in[8];
  const float* bv   = (const float*)d_in[9];
  const float* Wo   = (const float*)d_in[10];
  const float* bo   = (const float*)d_in[11];
  float* out = (float*)d_out;

  const size_t HSZ = (size_t)BS * HEADS * SEQ * DK;   // elements per bf16 tensor
  __bf16* Qh  = (__bf16*)d_ws;
  __bf16* KhT = Qh + HSZ;
  __bf16* Vh  = KhT + HSZ;
  __bf16* Oc  = Vh + HSZ;                             // 64 MB total workspace

  dim3 blk(256);
  dim3 gg(BS * SEQ / 128, D_MODEL / 64);              // (64, 16)
  gemm_bias_kernel<0, 0><<<gg, blk, 0, stream>>>(q, Wq, bq, Qh);
  gemm_bias_kernel<0, 1><<<gg, blk, 0, stream>>>(k, Wk, bk, KhT);
  gemm_bias_kernel<0, 0><<<gg, blk, 0, stream>>>(v, Wv, bv, Vh);

  dim3 ga(SEQ / 128, HEADS, BS);                      // (16, 16, 4)
  flash_attn_kernel<<<ga, blk, 0, stream>>>(Qh, KhT, Vh, mask, Oc);

  gemm_bias_kernel<1, 2><<<gg, blk, 0, stream>>>(Oc, Wo, bo, out);
}
